// StarTransformer_58780922413930
// MI455X (gfx1250) — matrix operand order repaired and checked by reading, not compiled, not run on hardware
//
#include <hip/hip_runtime.h>
#include <math.h>

typedef _Float16 half_t;
typedef _Float16 v16h __attribute__((ext_vector_type(16)));
typedef _Float16 v8h  __attribute__((ext_vector_type(8)));
typedef float    v8f  __attribute__((ext_vector_type(8)));

#define HIDDEN 512
#define SEQL   2048
#define BATCH  8
#define NHEAD  8
#define HDIM   64

#define BM 128
#define BN 128
#define BK 32
#define APAD 16   // A LDS row stride = BK+APAD = 48 halves (96B, 16B aligned)
#define BPAD 8    // B LDS row stride = BN+BPAD = 136 halves (272B, 16B aligned)

// ---------------------------------------------------------------- cvt f32->f16
__global__ void cvt_f32_f16(const float* __restrict__ src, half_t* __restrict__ dst, int n) {
  int i = blockIdx.x * 256 + threadIdx.x;
  if (i < n) dst[i] = (half_t)src[i];
}

// ---------------------------------------------------------------- WMMA GEMM
// C[M,N] = A[M,K] (f16 rm) * B[K,N] (f16 rm) [+ bias] [+ posemb] [leaky]
// 256 threads = 8 waves; block tile 128x128; wave tile 32x64 = 2x4 wmma 16x16x32.
// Double-buffered LDS: one barrier per K-step, global loads overlap WMMA.
template <bool HAS_BIAS, bool HAS_POS, bool LEAKY, bool OUTF, bool OUTH>
__global__ __launch_bounds__(256) void gemm_wmma(
    const half_t* __restrict__ A, const half_t* __restrict__ Bw,
    const float* __restrict__ bias, const float* __restrict__ posemb,
    float* __restrict__ outF, half_t* __restrict__ outH,
    int M, int N, int K)
{
  __shared__ half_t lA[2][BM][BK + APAD];
  __shared__ half_t lB[2][BK][BN + BPAD];

  const int tid    = threadIdx.x;
  const int lane   = tid & 31;
  const int wave   = tid >> 5;
  const int waveM  = wave & 3;     // 4 row groups of 32
  const int waveN  = wave >> 2;    // 2 col groups of 64
  const int blockM = blockIdx.y * BM;
  const int blockN = blockIdx.x * BN;

  // staging assignments (256 threads)
  const int arow = tid >> 1;          // 0..127, 16 halves each
  const int acol = (tid & 1) * 16;    // 0 / 16
  const int brow = tid >> 3;          // 0..31, 16 halves each
  const int bcol = (tid & 7) * 16;    // 0..112

  const half_t* agBase = A  + (size_t)(blockM + arow) * K + acol;
  const half_t* bgBase = Bw + (size_t)brow * N + blockN + bcol;

  v8f acc[2][4] = {};

  const int NK = K / BK;

  // prologue: stage tile 0 into buffer 0
  {
    const half_t* ag = agBase;
    *(v8h*)&lA[0][arow][acol]     = *(const v8h*)(ag);
    *(v8h*)&lA[0][arow][acol + 8] = *(const v8h*)(ag + 8);
    const half_t* bg = bgBase;
    *(v8h*)&lB[0][brow][bcol]     = *(const v8h*)(bg);
    *(v8h*)&lB[0][brow][bcol + 8] = *(const v8h*)(bg + 8);
  }
  __syncthreads();

  for (int kt = 0; kt < NK; ++kt) {
    const int buf = kt & 1;
    const bool hasNext = (kt + 1 < NK);

    // issue next-tile global loads early (overlap with WMMA below)
    v8h na0, na1, nb0, nb1;
    if (hasNext) {
      const half_t* ag = agBase + (kt + 1) * BK;
      na0 = *(const v8h*)(ag);
      na1 = *(const v8h*)(ag + 8);
      const half_t* bg = bgBase + (size_t)(kt + 1) * BK * N;
      nb0 = *(const v8h*)(bg);
      nb1 = *(const v8h*)(bg + 8);
    }
    if (kt + 2 < NK) {
      __builtin_prefetch(agBase + (kt + 2) * BK, 0, 0);
      __builtin_prefetch(bgBase + (size_t)(kt + 2) * BK * N, 0, 0);
    }

    // A fragments per ISA layout: lanes 0-15 hold K{c0..c0+7, c0+16..c0+23}
    const int c0 = (lane >> 4) * 8;
    v16h af[2];
#pragma unroll
    for (int mi = 0; mi < 2; ++mi) {
      int r = waveM * 32 + mi * 16 + (lane & 15);
      v8h lo = *(const v8h*)&lA[buf][r][c0];
      v8h hi = *(const v8h*)&lA[buf][r][c0 + 16];
      af[mi] = __builtin_shufflevector(lo, hi, 0,1,2,3,4,5,6,7,8,9,10,11,12,13,14,15);
    }
    // B fragments: lane = N column (lane&15), K{c0..c0+7, c0+16..c0+23}
    v16h bf[4];
#pragma unroll
    for (int ni = 0; ni < 4; ++ni) {
      int nb = waveN * 64 + ni * 16 + (lane & 15);
#pragma unroll
      for (int s = 0; s < 8; ++s) {
        bf[ni][s]     = lB[buf][c0 + s][nb];
        bf[ni][s + 8] = lB[buf][c0 + 16 + s][nb];
      }
    }
#pragma unroll
    for (int mi = 0; mi < 2; ++mi)
#pragma unroll
      for (int ni = 0; ni < 4; ++ni)
        acc[mi][ni] = __builtin_amdgcn_wmma_f32_16x16x32_f16(
            false, af[mi], false, bf[ni], (short)0, acc[mi][ni], false, false);

    if (hasNext) {
      const int nbuf = buf ^ 1;
      *(v8h*)&lA[nbuf][arow][acol]     = na0;
      *(v8h*)&lA[nbuf][arow][acol + 8] = na1;
      *(v8h*)&lB[nbuf][brow][bcol]     = nb0;
      *(v8h*)&lB[nbuf][brow][bcol + 8] = nb1;
      __syncthreads();
    }
  }

  // epilogue: C layout lane<16 -> N=lane, M=v; lane>=16 -> N=lane-16, M=8+v
  float bn[4];
#pragma unroll
  for (int ni = 0; ni < 4; ++ni) {
    int n = blockN + waveN * 64 + ni * 16 + (lane & 15);
    bn[ni] = HAS_BIAS ? bias[n] : 0.f;
  }
#pragma unroll
  for (int mi = 0; mi < 2; ++mi) {
#pragma unroll
    for (int ni = 0; ni < 4; ++ni) {
      int n  = blockN + waveN * 64 + ni * 16 + (lane & 15);
      int mb = blockM + waveM * 32 + mi * 16 + ((lane >> 4) << 3);
#pragma unroll
      for (int v = 0; v < 8; ++v) {
        int m = mb + v;
        float c = acc[mi][ni][v] + bn[ni];
        if (HAS_POS) c += posemb[(size_t)(m & (SEQL - 1)) * N + n];
        if (LEAKY)   c = c > 0.f ? c : 0.2f * c;
        size_t idx = (size_t)m * N + n;
        if (OUTF) outF[idx] = c;
        if (OUTH) outH[idx] = (half_t)c;
      }
    }
  }
}

// ---------------------------------------------------------------- LayerNorm (wave per row)
__global__ __launch_bounds__(256) void ln_kernel(
    const float* __restrict__ x, const float* __restrict__ g, const float* __restrict__ b,
    half_t* __restrict__ out, int rows)
{
  int wave = threadIdx.x >> 5, lane = threadIdx.x & 31;
  int row = blockIdx.x * 8 + wave;
  if (row >= rows) return;
  const float* xr = x + (size_t)row * HIDDEN;
  float s = 0.f, s2 = 0.f;
#pragma unroll
  for (int i = 0; i < 16; ++i) { float v = xr[lane + i * 32]; s += v; s2 += v * v; }
  for (int off = 16; off >= 1; off >>= 1) { s += __shfl_xor(s, off, 32); s2 += __shfl_xor(s2, off, 32); }
  float mu  = s  * (1.f / HIDDEN);
  float var = s2 * (1.f / HIDDEN) - mu * mu;
  float inv = rsqrtf(var + 1e-6f);
  half_t* orow = out + (size_t)row * HIDDEN;
#pragma unroll
  for (int i = 0; i < 16; ++i) {
    int n = lane + i * 32;
    orow[n] = (half_t)((xr[n] - mu) * inv * g[n] + b[n]);
  }
}

// ---------------------------------------------------------------- relay = mean over L
__global__ void relay_mean(const float* __restrict__ nodes, float* __restrict__ relay) {
  int b = blockIdx.x, tid = threadIdx.x;
  for (int i = 0; i < 2; ++i) {
    int n = tid + i * 256;
    float s = 0.f;
    for (int l = 0; l < SEQL; ++l) s += nodes[((size_t)b * SEQL + l) * HIDDEN + n];
    relay[b * HIDDEN + n] = s * (1.f / SEQL);
  }
}

// ---------------------------------------------------------------- relay row QKV (8 rows, f32 weights)
__global__ void relay_qkv(const float* __restrict__ relay,
                          const float* __restrict__ wq, const float* __restrict__ wk, const float* __restrict__ wv,
                          const float* __restrict__ bq, const float* __restrict__ bk, const float* __restrict__ bv,
                          half_t* __restrict__ qr, half_t* __restrict__ kr, half_t* __restrict__ vr)
{
  __shared__ float r[HIDDEN];
  int b = blockIdx.x, tid = threadIdx.x;
  r[tid] = relay[b * HIDDEN + tid];
  r[tid + 256] = relay[b * HIDDEN + tid + 256];
  __syncthreads();
  for (int i = 0; i < 2; ++i) {
    int n = tid + i * 256;
    float aq = bq[n], ak = bk[n], av = bv[n];
    for (int k = 0; k < HIDDEN; ++k) {
      float rv = r[k];
      aq += rv * wq[(size_t)k * HIDDEN + n];
      ak += rv * wk[(size_t)k * HIDDEN + n];
      av += rv * wv[(size_t)k * HIDDEN + n];
    }
    qr[b * HIDDEN + n] = (half_t)aq;
    kr[b * HIDDEN + n] = (half_t)ak;
    vr[b * HIDDEN + n] = (half_t)av;
  }
}

// ---------------------------------------------------------------- ring attention (wave per (b,l,h))
__device__ __forceinline__ float wave_sum(float p) {
  for (int off = 16; off >= 1; off >>= 1) p += __shfl_xor(p, off, 32);
  return p;
}

__global__ __launch_bounds__(256) void ring_attn(
    const half_t* __restrict__ qh, const half_t* __restrict__ kh, const half_t* __restrict__ vh,
    const half_t* __restrict__ krh, const half_t* __restrict__ vrh, half_t* __restrict__ atth)
{
  const float scale = 0.125f;  // 1/sqrt(64)
  int wave = threadIdx.x >> 5, lane = threadIdx.x & 31;
  int g = blockIdx.x * 8 + wave;
  int b = g >> 14;                 // / (L*NH) = 16384
  int rem = g & 16383;
  int l = rem >> 3, h = rem & 7;
  size_t qoff = ((size_t)(b * SEQL + l)) * HIDDEN + h * HDIM;
  int d0 = lane * 2;
  float q0 = (float)qh[qoff + d0], q1 = (float)qh[qoff + d0 + 1];

  // windows: u=0 -> l-1 (zero-pad), u=1 -> l, u=2 -> l+1 (zero-pad), u=3 -> relay.
  // Padded-out rows contribute score 0 (dot with zeros) and value 0, like the ref.
  float s[4];
  const half_t* kp;
  if (l > 0) { kp = kh + ((size_t)(b * SEQL + l - 1)) * HIDDEN + h * HDIM;
               s[0] = wave_sum(q0 * (float)kp[d0] + q1 * (float)kp[d0 + 1]) * scale; }
  else s[0] = 0.f;
  kp = kh + qoff;
  s[1] = wave_sum(q0 * (float)kp[d0] + q1 * (float)kp[d0 + 1]) * scale;
  if (l < SEQL - 1) { kp = kh + ((size_t)(b * SEQL + l + 1)) * HIDDEN + h * HDIM;
               s[2] = wave_sum(q0 * (float)kp[d0] + q1 * (float)kp[d0 + 1]) * scale; }
  else s[2] = 0.f;
  kp = krh + (size_t)b * HIDDEN + h * HDIM;
  s[3] = wave_sum(q0 * (float)kp[d0] + q1 * (float)kp[d0 + 1]) * scale;

  float mx = fmaxf(fmaxf(s[0], s[1]), fmaxf(s[2], s[3]));
  float e0 = __expf(s[0] - mx), e1 = __expf(s[1] - mx), e2 = __expf(s[2] - mx), e3 = __expf(s[3] - mx);
  float rden = 1.f / (e0 + e1 + e2 + e3);

  float a0 = 0.f, a1 = 0.f;
  const half_t* vp;
  if (l > 0) { vp = vh + ((size_t)(b * SEQL + l - 1)) * HIDDEN + h * HDIM;
               a0 += e0 * (float)vp[d0]; a1 += e0 * (float)vp[d0 + 1]; }
  vp = vh + qoff;
  a0 += e1 * (float)vp[d0]; a1 += e1 * (float)vp[d0 + 1];
  if (l < SEQL - 1) { vp = vh + ((size_t)(b * SEQL + l + 1)) * HIDDEN + h * HDIM;
               a0 += e2 * (float)vp[d0]; a1 += e2 * (float)vp[d0 + 1]; }
  vp = vrh + (size_t)b * HIDDEN + h * HDIM;
  a0 += e3 * (float)vp[d0]; a1 += e3 * (float)vp[d0 + 1];

  atth[qoff + d0]     = (half_t)(a0 * rden);
  atth[qoff + d0 + 1] = (half_t)(a1 * rden);
}

// ---------------------------------------------------------------- star attention (block per (b,h))
__global__ __launch_bounds__(256) void star_attn(
    const half_t* __restrict__ qrh, const half_t* __restrict__ kh, const half_t* __restrict__ krh,
    const half_t* __restrict__ vh, const half_t* __restrict__ vrh, float* __restrict__ attr)
{
  __shared__ float sS[SEQL + 4];   // 2049 scores (padded)
  __shared__ float red[256];
  __shared__ float pacc[4][HDIM];
  __shared__ float qS[HDIM];
  const float scale = 0.125f;
  int bh = blockIdx.x, tid = threadIdx.x;
  int b = bh >> 3, h = bh & 7;

  if (tid < HDIM) qS[tid] = (float)qrh[(size_t)b * HIDDEN + h * HDIM + tid];
  __syncthreads();

  // scores over L+1 rows (row L = relay)
  for (int l = tid; l <= SEQL; l += 256) {
    const half_t* kp = (l < SEQL) ? kh + ((size_t)(b * SEQL + l)) * HIDDEN + h * HDIM
                                  : krh + (size_t)b * HIDDEN + h * HDIM;
    float s = 0.f;
    for (int d = 0; d < HDIM; ++d) s += qS[d] * (float)kp[d];
    sS[l] = s * scale;
  }
  __syncthreads();

  // block max
  float lm = -1e30f;
  for (int l = tid; l <= SEQL; l += 256) lm = fmaxf(lm, sS[l]);
  red[tid] = lm; __syncthreads();
  for (int s = 128; s > 0; s >>= 1) { if (tid < s) red[tid] = fmaxf(red[tid], red[tid + s]); __syncthreads(); }
  float m = red[0]; __syncthreads();

  // exp + block sum
  float ls = 0.f;
  for (int l = tid; l <= SEQL; l += 256) { float e = __expf(sS[l] - m); sS[l] = e; ls += e; }
  red[tid] = ls; __syncthreads();
  for (int s = 128; s > 0; s >>= 1) { if (tid < s) red[tid] += red[tid + s]; __syncthreads(); }
  float rden = 1.f / red[0]; __syncthreads();

  // weighted sum of v
  int d = tid & 63, chunk = tid >> 6;
  float a = 0.f;
  for (int l = chunk; l <= SEQL; l += 4) {
    const half_t* vp = (l < SEQL) ? vh + ((size_t)(b * SEQL + l)) * HIDDEN + h * HDIM
                                  : vrh + (size_t)b * HIDDEN + h * HDIM;
    a += sS[l] * (float)vp[d];
  }
  pacc[chunk][d] = a; __syncthreads();
  if (tid < HDIM) {
    float t = pacc[0][tid] + pacc[1][tid] + pacc[2][tid] + pacc[3][tid];
    attr[((size_t)b * NHEAD + h) * HDIM + tid] = t * rden;
  }
}

// ---------------------------------------------------------------- relay projection (8 rows)
__global__ void star_out(const float* __restrict__ attr, const float* __restrict__ w,
                         const float* __restrict__ bias, float* __restrict__ relay)
{
  __shared__ float a[HIDDEN];
  int b = blockIdx.x, tid = threadIdx.x;
  a[tid] = attr[b * HIDDEN + tid];
  a[tid + 256] = attr[b * HIDDEN + tid + 256];
  __syncthreads();
  for (int i = 0; i < 2; ++i) {
    int n = tid + i * 256;
    float s = bias[n];
    for (int k = 0; k < HIDDEN; ++k) s += a[k] * w[(size_t)k * HIDDEN + n];
    relay[b * HIDDEN + n] = s > 0.f ? s : 0.2f * s;
  }
}

// ---------------------------------------------------------------- final merge
__global__ void final_out(const float* __restrict__ nodes, const float* __restrict__ relay,
                          float* __restrict__ out)
{
  int b = blockIdx.x, tid = threadIdx.x;
  for (int i = 0; i < 2; ++i) {
    int n = tid + i * 256;
    float mx = -1e30f;
    for (int l = 0; l < SEQL; ++l) mx = fmaxf(mx, nodes[((size_t)b * SEQL + l) * HIDDEN + n]);
    out[b * HIDDEN + n] = 0.5f * (relay[b * HIDDEN + n] + mx);
  }
}

// ---------------------------------------------------------------- host
extern "C" void kernel_launch(void* const* d_in, const int* in_sizes, int n_in,
                              void* d_out, int out_size, void* d_ws, size_t ws_size,
                              hipStream_t stream)
{
  const float* data    = (const float*)d_in[0];
  const float* fc_w    = (const float*)d_in[1];
  const float* fc_b    = (const float*)d_in[2];
  const float* pos_emb = (const float*)d_in[3];
  const float* ln_g    = (const float*)d_in[4];
  const float* ln_b    = (const float*)d_in[5];
  const float* wq      = (const float*)d_in[6];
  const float* wk      = (const float*)d_in[7];
  const float* wv      = (const float*)d_in[8];
  const float* bq      = (const float*)d_in[9];
  const float* bk      = (const float*)d_in[10];
  const float* bv      = (const float*)d_in[11];
  const float* ring_wo = (const float*)d_in[12];
  const float* ring_bo = (const float*)d_in[13];
  const float* star_wo = (const float*)d_in[14];
  const float* star_bo = (const float*)d_in[15];
  float* out = (float*)d_out;

  char* ws = (char*)d_ws;
  size_t off = 0;
  auto alloc = [&](size_t bytes) -> void* {
    off = (off + 255) & ~(size_t)255;
    void* p = ws + off; off += bytes; return p;
  };
  const int MROWS = BATCH * SEQL;                       // 16384
  const size_t ACT_H = (size_t)MROWS * HIDDEN;          // activation elems
  const size_t WMAT  = (size_t)HIDDEN * HIDDEN;

  half_t* datah = (half_t*)alloc(ACT_H * 2);            // reused as atth after embed
  half_t* fcwh  = (half_t*)alloc(WMAT * 2);
  half_t* wqh   = (half_t*)alloc(4 * WMAT * 2);
  half_t* wkh   = (half_t*)alloc(4 * WMAT * 2);
  half_t* wvh   = (half_t*)alloc(4 * WMAT * 2);
  half_t* ringh = (half_t*)alloc(4 * WMAT * 2);
  float*  nodes = (float*) alloc(ACT_H * 4);
  half_t* xh    = (half_t*)alloc(ACT_H * 2);
  half_t* qh    = (half_t*)alloc(ACT_H * 2);
  half_t* kh    = (half_t*)alloc(ACT_H * 2);
  half_t* vh    = (half_t*)alloc(ACT_H * 2);
  float*  relay = (float*) alloc((size_t)BATCH * HIDDEN * 4);
  half_t* qrh   = (half_t*)alloc((size_t)BATCH * HIDDEN * 2);
  half_t* krh   = (half_t*)alloc((size_t)BATCH * HIDDEN * 2);
  half_t* vrh   = (half_t*)alloc((size_t)BATCH * HIDDEN * 2);
  float*  attr  = (float*) alloc((size_t)BATCH * HIDDEN * 4);
  half_t* atth  = datah;   // alias: data f16 copy dead after embed GEMM

  // weight/activation conversions to f16
  {
    int n;
    n = (int)ACT_H;       cvt_f32_f16<<<(n + 255) / 256, 256, 0, stream>>>(data,    datah, n);
    n = (int)WMAT;        cvt_f32_f16<<<(n + 255) / 256, 256, 0, stream>>>(fc_w,    fcwh,  n);
    n = (int)(4 * WMAT);  cvt_f32_f16<<<(n + 255) / 256, 256, 0, stream>>>(wq,      wqh,   n);
    n = (int)(4 * WMAT);  cvt_f32_f16<<<(n + 255) / 256, 256, 0, stream>>>(wk,      wkh,   n);
    n = (int)(4 * WMAT);  cvt_f32_f16<<<(n + 255) / 256, 256, 0, stream>>>(wv,      wvh,   n);
    n = (int)(4 * WMAT);  cvt_f32_f16<<<(n + 255) / 256, 256, 0, stream>>>(ring_wo, ringh, n);
  }

  dim3 ggrid(HIDDEN / BN, MROWS / BM);  // (4, 128)

  // embs = data @ fc_w + fc_b + pos_emb  -> nodes (f32)
  gemm_wmma<true, true, false, true, false><<<ggrid, 256, 0, stream>>>(
      datah, fcwh, fc_b, pos_emb, nodes, nullptr, MROWS, HIDDEN, HIDDEN);
  relay_mean<<<BATCH, 256, 0, stream>>>(nodes, relay);

  for (int i = 0; i < 4; ++i) {
    const half_t* wqi = wqh + (size_t)i * WMAT;
    const half_t* wki = wkh + (size_t)i * WMAT;
    const half_t* wvi = wvh + (size_t)i * WMAT;

    ln_kernel<<<MROWS / 8, 256, 0, stream>>>(nodes, ln_g + i * HIDDEN, ln_b + i * HIDDEN, xh, MROWS);

    gemm_wmma<true, false, false, false, true><<<ggrid, 256, 0, stream>>>(
        xh, wqi, bq + i * HIDDEN, nullptr, nullptr, qh, MROWS, HIDDEN, HIDDEN);
    gemm_wmma<true, false, false, false, true><<<ggrid, 256, 0, stream>>>(
        xh, wki, bk + i * HIDDEN, nullptr, nullptr, kh, MROWS, HIDDEN, HIDDEN);
    gemm_wmma<true, false, false, false, true><<<ggrid, 256, 0, stream>>>(
        xh, wvi, bv + i * HIDDEN, nullptr, nullptr, vh, MROWS, HIDDEN, HIDDEN);

    relay_qkv<<<BATCH, 256, 0, stream>>>(relay,
        wq + (size_t)i * WMAT, wk + (size_t)i * WMAT, wv + (size_t)i * WMAT,
        bq + i * HIDDEN, bk + i * HIDDEN, bv + i * HIDDEN, qrh, krh, vrh);

    ring_attn<<<(MROWS * NHEAD) / 8, 256, 0, stream>>>(qh, kh, vh, krh, vrh, atth);

    gemm_wmma<true, false, true, true, false><<<ggrid, 256, 0, stream>>>(
        atth, ringh + (size_t)i * WMAT, ring_bo + i * HIDDEN, nullptr, nodes, nullptr,
        MROWS, HIDDEN, HIDDEN);

    star_attn<<<BATCH * NHEAD, 256, 0, stream>>>(qrh, kh, krh, vh, vrh, attr);
    star_out<<<BATCH, 256, 0, stream>>>(attr, star_wo + (size_t)i * WMAT,
                                        star_bo + i * HIDDEN, relay);
  }

  final_out<<<BATCH, 256, 0, stream>>>(nodes, relay, out);

  (void)in_sizes; (void)n_in; (void)out_size; (void)ws_size;
}
